// GATE11a_53111565582535
// MI455X (gfx1250) — compile-verified, matrix operand order checked by reading, not run on hardware
//
#include <hip/hip_runtime.h>
#include <hip/hip_bf16.h>

// ---------------------------------------------------------------------------
// GATv2 pipeline for MI455X (gfx1250), wave32 + v_wmma_f32_16x16x32_bf16.
// Weights pre-packed to bf16 in WMMA fragment order (contiguous b128 loads);
// activations read as aligned float4 chunks and converted in-register.
// ---------------------------------------------------------------------------

#define NODES 32768
#define EDGES 262144
#define GR    128
#define NEG_SLOPE 0.2f
#define EPS_BN 1e-5f

typedef __bf16 bf16_t;
typedef __attribute__((ext_vector_type(16))) __bf16 v16bf;
typedef __attribute__((ext_vector_type(8)))  float  v8f;

__device__ __forceinline__ float4 ld4(const float* p) { return *(const float4*)p; }

// write 8 fp32 values (two float4) into fragment elements [base, base+8)
__device__ __forceinline__ void put8(v16bf& a, int base, float4 x, float4 y) {
  a[base + 0] = (bf16_t)x.x; a[base + 1] = (bf16_t)x.y;
  a[base + 2] = (bf16_t)x.z; a[base + 3] = (bf16_t)x.w;
  a[base + 4] = (bf16_t)y.x; a[base + 5] = (bf16_t)y.y;
  a[base + 6] = (bf16_t)y.z; a[base + 7] = (bf16_t)y.w;
}
__device__ __forceinline__ void zero8(v16bf& a, int base) {
#pragma unroll
  for (int e = 0; e < 8; ++e) a[base + e] = (bf16_t)0.f;
}

__device__ __forceinline__ void atomicMaxF(float* addr, float val) {
  unsigned int* ua = (unsigned int*)addr;
  unsigned int old = *ua;
  while (__uint_as_float(old) < val) {
    unsigned int assumed = old;
    old = atomicCAS(ua, assumed, __float_as_uint(val));
    if (old == assumed) break;
  }
}

// ---------------------------------------------------------------------------
// Pre-pack a K x N fp32 weight matrix into bf16 WMMA B-fragments:
// element e of lane's fragment for tile (kt,nt):
//   k = kt*32 + (e<8 ? 8*lh+e : 16+8*lh+(e-8)),  col = nt*16 + (lane&15)
// stored contiguously so a lane reads its 16 bf16 (32B) in one shot.
// ---------------------------------------------------------------------------
__global__ void pack_w(const float* __restrict__ B, int K, int N,
                       bf16_t* __restrict__ out) {
  const int i = blockIdx.x * blockDim.x + threadIdx.x;
  const int Kpad = (K + 31) & ~31;
  if (i >= Kpad * N) return;
  const int e = i & 15;
  const int lane = (i >> 4) & 31;
  const int tile = i >> 9;
  const int ntiles = N >> 4;
  const int kt = tile / ntiles, nt = tile - kt * ntiles;
  const int lh = lane >> 4;
  const int k = kt * 32 + (e < 8 ? lh * 8 + e : 16 + lh * 8 + (e - 8));
  const int col = nt * 16 + (lane & 15);
  out[i] = (k < K) ? (bf16_t)B[(size_t)k * N + col] : (bf16_t)0.f;
}

// ---------------------------------------------------------------------------
// Tiled GEMM: C[M,N] = act(A[M,K] * B + bias). One wave -> 16x64 output
// (4 n-tiles reuse one A fragment per K-step). A read as aligned float4.
// ---------------------------------------------------------------------------
template <int ACT>
__global__ void gemm_wmma(const float* __restrict__ A, int lda,
                          const bf16_t* __restrict__ Bp,
                          const float* __restrict__ bias,
                          float* __restrict__ C, int ldc,
                          int M, int N, int K) {
  const int ntiles = N >> 4, ngroups = ntiles >> 2;
  const int wave = blockIdx.x * (blockDim.x >> 5) + (threadIdx.x >> 5);
  const int lane = threadIdx.x & 31;
  const int l16 = lane & 15, lh = lane >> 4;
  const int mt = wave / ngroups, ng = wave - mt * ngroups;
  if (mt * 16 >= M) return;
  const float* Arow = A + (size_t)(mt * 16 + l16) * lda;
  const v16bf* bp = (const v16bf*)Bp;
  v8f acc[4] = {{}, {}, {}, {}};
  for (int k0 = 0; k0 < K; k0 += 32) {
    v16bf a;
    const int kA = k0 + lh * 8;        // chunk for fragment elems 0..7
    const int kA2 = k0 + 16 + lh * 8;  // chunk for fragment elems 8..15
    if (kA  < K) put8(a, 0, ld4(Arow + kA),  ld4(Arow + kA + 4));  else zero8(a, 0);
    if (kA2 < K) put8(a, 8, ld4(Arow + kA2), ld4(Arow + kA2 + 4)); else zero8(a, 8);
    const v16bf* bt = bp + ((size_t)(k0 >> 5) * ntiles + ng * 4) * 32 + lane;
#pragma unroll
    for (int t = 0; t < 4; ++t) {
      const v16bf b = bt[t * 32];
      acc[t] = __builtin_amdgcn_wmma_f32_16x16x32_bf16(false, a, false, b,
                                                       (short)0, acc[t], false, false);
    }
  }
#pragma unroll
  for (int t = 0; t < 4; ++t) {
    const int col = ng * 64 + t * 16 + l16;
    const float bv = bias[col];
#pragma unroll
    for (int v = 0; v < 8; ++v) {
      const int m = mt * 16 + v + 8 * lh;
      float c = acc[t][v] + bv;
      if (ACT) c = fmaxf(c, 0.f);
      C[(size_t)m * ldc + col] = c;
    }
  }
}

// ---------------------------------------------------------------------------
// Gather-GEMM: A row r = [h[src[r]] (256) | h[dst[r]] (256) | eattr[r] (edim)].
// Segment boundaries (256/512) are 8-aligned so each 8-float chunk lives in
// exactly one segment -> still aligned float4 loads.
// ---------------------------------------------------------------------------
__device__ __forceinline__ float4 ecat4(const float* __restrict__ htab,
                                        const float* __restrict__ eattr,
                                        int edim, int sr, int dr, int r, int c) {
  if (c < 256) return ld4(htab + (size_t)sr * 256 + c);
  if (c < 512) return ld4(htab + (size_t)dr * 256 + (c - 256));
  return ld4(eattr + (size_t)r * edim + (c - 512));
}

template <int ACT>
__global__ void gemm_ecat_wmma(const float* __restrict__ htab,
                               const float* __restrict__ eattr, int edim,
                               const int* __restrict__ src,
                               const int* __restrict__ dst,
                               const bf16_t* __restrict__ Bp,
                               const float* __restrict__ bias,
                               float* __restrict__ C, int ldc,
                               int M, int N, int K) {
  const int ntiles = N >> 4, ngroups = ntiles >> 2;
  const int wave = blockIdx.x * (blockDim.x >> 5) + (threadIdx.x >> 5);
  const int lane = threadIdx.x & 31;
  const int l16 = lane & 15, lh = lane >> 4;
  const int mt = wave / ngroups, ng = wave - mt * ngroups;
  if (mt * 16 >= M) return;
  const int r = mt * 16 + l16;
  const int sr = src[r], dr = dst[r];
  const v16bf* bp = (const v16bf*)Bp;
  v8f acc[4] = {{}, {}, {}, {}};
  for (int k0 = 0; k0 < K; k0 += 32) {
    v16bf a;
    const int kA = k0 + lh * 8;
    const int kA2 = k0 + 16 + lh * 8;
    if (kA < K)
      put8(a, 0, ecat4(htab, eattr, edim, sr, dr, r, kA),
                 ecat4(htab, eattr, edim, sr, dr, r, kA + 4));
    else zero8(a, 0);
    if (kA2 < K)
      put8(a, 8, ecat4(htab, eattr, edim, sr, dr, r, kA2),
                 ecat4(htab, eattr, edim, sr, dr, r, kA2 + 4));
    else zero8(a, 8);
    const v16bf* bt = bp + ((size_t)(k0 >> 5) * ntiles + ng * 4) * 32 + lane;
#pragma unroll
    for (int t = 0; t < 4; ++t) {
      const v16bf b = bt[t * 32];
      acc[t] = __builtin_amdgcn_wmma_f32_16x16x32_bf16(false, a, false, b,
                                                       (short)0, acc[t], false, false);
    }
  }
#pragma unroll
  for (int t = 0; t < 4; ++t) {
    const int col = ng * 64 + t * 16 + l16;
    const float bv = bias[col];
#pragma unroll
    for (int v = 0; v < 8; ++v) {
      const int m = mt * 16 + v + 8 * lh;
      float c = acc[t][v] + bv;
      if (ACT) c = fmaxf(c, 0.f);
      C[(size_t)m * ldc + col] = c;
    }
  }
}

// ---------------------------------------------------------------------------
// Fused GATv2 attention logits: one wave = 16 edges. The 4 K-step A-fragments
// of ea (K=128) are loaded ONCE and reused across all 16 head x tile WMMAs of
// ee = ea @ We. Gather addresses for xl[src]/xr[dst] are kept as 32-bit
// element offsets (sm*256 < 2^23) so the compiler folds base+voffset instead
// of holding 16 spilled 64-bit pointers.
// ---------------------------------------------------------------------------
__global__ void gat_alpha_kernel(const float* __restrict__ ea,      // E x 128
                                 const float* __restrict__ ea_mean, // 128
                                 const bf16_t* __restrict__ Wep,    // packed 128x256
                                 const float* __restrict__ xl,
                                 const float* __restrict__ xr,
                                 const int* __restrict__ src,
                                 const int* __restrict__ dst,
                                 const float* __restrict__ att,     // 4 x 64
                                 float* __restrict__ alpha_raw,     // (E+N) x 4
                                 float* __restrict__ amax,          // N x 4
                                 int E, int Nn) {
  const int wave = blockIdx.x * (blockDim.x >> 5) + (threadIdx.x >> 5);
  const int lane = threadIdx.x & 31;
  const int l16 = lane & 15, lh = lane >> 4;
  const int r0 = wave * 16;
  if (r0 >= E + Nn) return;
  const int erow = r0 + l16;
  const float* arow = (erow < E) ? (ea + (size_t)erow * 128) : ea_mean;
  const v16bf* bp = (const v16bf*)Wep;

  v16bf af[4];                         // K = 128 -> 4 fragments, loaded once
#pragma unroll
  for (int ks = 0; ks < 4; ++ks) {
    const int kA = ks * 32 + lh * 8;
    const int kA2 = ks * 32 + 16 + lh * 8;
    put8(af[ks], 0, ld4(arow + kA),  ld4(arow + kA + 4));
    put8(af[ks], 8, ld4(arow + kA2), ld4(arow + kA2 + 4));
  }
  int sOff[8], dOff[8], dRow[8];       // 32-bit gather offsets (elements)
#pragma unroll
  for (int v = 0; v < 8; ++v) {
    const int rr = r0 + v + 8 * lh;
    const int s = (rr < E) ? src[rr] : (rr - E);
    const int d = (rr < E) ? dst[rr] : (rr - E);
    sOff[v] = s * 256;
    dOff[v] = d * 256;
    dRow[v] = d;
  }
  for (int head = 0; head < 4; ++head) {
    float aacc[8];
#pragma unroll
    for (int v = 0; v < 8; ++v) aacc[v] = 0.f;
    for (int ntl = 0; ntl < 4; ++ntl) {
      const int nt = head * 4 + ntl;
      const int col = nt * 16 + l16;
      v8f acc = {};
#pragma unroll
      for (int ks = 0; ks < 4; ++ks) {
        const v16bf b = bp[((size_t)ks * 16 + nt) * 32 + lane];
        acc = __builtin_amdgcn_wmma_f32_16x16x32_bf16(false, af[ks], false, b,
                                                      (short)0, acc, false, false);
      }
      const float attv = att[head * 64 + ntl * 16 + l16];
#pragma unroll
      for (int v = 0; v < 8; ++v) {
        float z = acc[v] + xl[sOff[v] + col] + xr[dOff[v] + col];
        z = (z > 0.f) ? z : z * NEG_SLOPE;
        aacc[v] += z * attv;
      }
    }
#pragma unroll
    for (int v = 0; v < 8; ++v) {      // reduce over 16 column-lanes per half
      float s = aacc[v];
      for (int off = 1; off < 16; off <<= 1) s += __shfl_xor(s, off, 32);
      aacc[v] = s;
    }
    if (l16 == 0) {
#pragma unroll
      for (int v = 0; v < 8; ++v) {
        const int rr = r0 + v + 8 * lh;
        alpha_raw[(size_t)rr * 4 + head] = aacc[v];
        atomicMaxF(&amax[(size_t)dRow[v] * 4 + head], aacc[v]);
      }
    }
  }
}

__global__ void softmax_edge(const float* __restrict__ alpha_raw,
                             const float* __restrict__ amax,
                             const int* __restrict__ dst,
                             float* __restrict__ alpha_exp,
                             float* __restrict__ denom, int E, int Nn) {
  const int i = blockIdx.x * blockDim.x + threadIdx.x;
  if (i >= (E + Nn) * 4) return;
  const int r = i >> 2, h = i & 3;
  const int d = (r < E) ? dst[r] : (r - E);
  const float a = expf(alpha_raw[i] - amax[d * 4 + h]);
  alpha_exp[i] = a;
  atomicAdd(&denom[d * 4 + h], a);
}

__global__ void scatter_msg(const float* __restrict__ alpha_exp,
                            const float* __restrict__ denom,
                            const float* __restrict__ xl,
                            const int* __restrict__ src,
                            const int* __restrict__ dst,
                            float* __restrict__ out, int E, int Nn) {
  const int i = blockIdx.x * blockDim.x + threadIdx.x;
  if (i >= (E + Nn) * 256) return;
  const int r = i >> 8, c = i & 255;
  const int s = (r < E) ? src[r] : (r - E);
  const int d = (r < E) ? dst[r] : (r - E);
  const int h = c >> 6;
  const float a = alpha_exp[r * 4 + h] / (denom[d * 4 + h] + 1e-16f);
  atomicAdd(&out[(size_t)d * 256 + c], xl[(size_t)s * 256 + c] * a);
}

__global__ void bias_relu(float* __restrict__ X, const float* __restrict__ bias,
                          int rows, int cols) {
  const int i = blockIdx.x * blockDim.x + threadIdx.x;
  if (i >= rows * cols) return;
  X[i] = fmaxf(X[i] + bias[i % cols], 0.f);
}

__global__ void fill_f32(float* __restrict__ p, float v, int n) {
  const int i = blockIdx.x * blockDim.x + threadIdx.x;
  if (i < n) p[i] = v;
}

__global__ void col_mean(const float* __restrict__ X, int rows,
                         float* __restrict__ mean) {
  __shared__ float s[128];
  for (int c = threadIdx.x; c < 128; c += blockDim.x) s[c] = 0.f;
  __syncthreads();
  const int per = (rows + gridDim.x - 1) / gridDim.x;
  const int r0 = blockIdx.x * per;
  int r1 = r0 + per; if (r1 > rows) r1 = rows;
  for (size_t idx = (size_t)r0 * 128 + threadIdx.x; idx < (size_t)r1 * 128;
       idx += blockDim.x)
    atomicAdd(&s[(int)(idx & 127)], X[idx]);
  __syncthreads();
  for (int c = threadIdx.x; c < 128; c += blockDim.x)
    atomicAdd(&mean[c], s[c] / (float)rows);
}

__global__ void bn_stats(const float* __restrict__ X, int rows, int cols,
                         float* __restrict__ mv) {
  const int c = blockIdx.x;
  float s = 0.f, s2 = 0.f;
  for (int r = threadIdx.x; r < rows; r += blockDim.x) {
    const float x = X[(size_t)r * cols + c];
    s += x; s2 += x * x;
  }
  __shared__ float sh[512];
  sh[threadIdx.x] = s; sh[256 + threadIdx.x] = s2;
  __syncthreads();
  for (int st = 128; st > 0; st >>= 1) {
    if ((int)threadIdx.x < st) {
      sh[threadIdx.x] += sh[threadIdx.x + st];
      sh[256 + threadIdx.x] += sh[256 + threadIdx.x + st];
    }
    __syncthreads();
  }
  if (threadIdx.x == 0) {
    const float m = sh[0] / (float)rows;
    mv[c] = m;
    mv[cols + c] = sh[256] / (float)rows - m * m;
  }
}

__global__ void bn_apply(float* __restrict__ X, int rows, int cols,
                         const float* __restrict__ mv,
                         const float* __restrict__ g,
                         const float* __restrict__ b) {
  const size_t i = (size_t)blockIdx.x * blockDim.x + threadIdx.x;
  if (i >= (size_t)rows * cols) return;
  const int c = (int)(i % cols);
  X[i] = (X[i] - mv[c]) * rsqrtf(mv[cols + c] + EPS_BN) * g[c] + b[c];
}

__global__ void pool_kernel(const float* __restrict__ h2,
                            const int* __restrict__ batch,
                            float* __restrict__ pooled, int Nn) {
  const int i = blockIdx.x * blockDim.x + threadIdx.x;
  if (i >= Nn * 256) return;
  const int n = i >> 8, c = i & 255;
  atomicAdd(&pooled[(size_t)batch[n] * 256 + c], h2[i]);
}

__global__ void fc2_kernel(const float* __restrict__ fin, const float* __restrict__ W,
                           const float* __restrict__ b, float* __restrict__ out, int G) {
  const int g = blockIdx.x * blockDim.x + threadIdx.x;
  if (g >= G) return;
  float s = b[0];
#pragma unroll 8
  for (int k = 0; k < 64; ++k) s += fin[g * 64 + k] * W[k];
  out[g] = s;
}

// ---------------------------------------------------------------------------
static inline void run_pack(const float* B, int K, int N, bf16_t* out, hipStream_t s) {
  const int Kpad = (K + 31) & ~31;
  pack_w<<<(Kpad * N + 255) / 256, 256, 0, s>>>(B, K, N, out);
}
static inline void run_gemm(const float* A, int lda, const bf16_t* Bp,
                            const float* bias, float* C, int ldc,
                            int M, int N, int K, int act, hipStream_t s) {
  const int blocks = ((M / 16) * (N / 64) + 7) / 8;
  if (act) gemm_wmma<1><<<blocks, 256, 0, s>>>(A, lda, Bp, bias, C, ldc, M, N, K);
  else     gemm_wmma<0><<<blocks, 256, 0, s>>>(A, lda, Bp, bias, C, ldc, M, N, K);
}
static inline void run_ecat(const float* htab, const float* eattr, int edim,
                            const int* src, const int* dst, const bf16_t* Bp,
                            const float* bias, float* C, int ldc,
                            int M, int N, int K, int act, hipStream_t s) {
  const int blocks = ((M / 16) * (N / 64) + 7) / 8;
  if (act) gemm_ecat_wmma<1><<<blocks, 256, 0, s>>>(htab, eattr, edim, src, dst,
                                                    Bp, bias, C, ldc, M, N, K);
  else     gemm_ecat_wmma<0><<<blocks, 256, 0, s>>>(htab, eattr, edim, src, dst,
                                                    Bp, bias, C, ldc, M, N, K);
}
static inline void fill(float* p, float v, int n, hipStream_t s) {
  fill_f32<<<(n + 255) / 256, 256, 0, s>>>(p, v, n);
}

extern "C" void kernel_launch(void* const* d_in, const int* in_sizes, int n_in,
                              void* d_out, int out_size, void* d_ws, size_t ws_size,
                              hipStream_t stream) {
  const float* x      = (const float*)d_in[0];
  const int*   eidx   = (const int*)d_in[1];
  const float* eattr0 = (const float*)d_in[2];
  const int*   batch  = (const int*)d_in[3];
  const float* nt_W1 = (const float*)d_in[4];  const float* nt_b1 = (const float*)d_in[5];
  const float* nt_W2 = (const float*)d_in[6];  const float* nt_b2 = (const float*)d_in[7];
  const float* e1_W1 = (const float*)d_in[8];  const float* e1_b1 = (const float*)d_in[9];
  const float* e1_W2 = (const float*)d_in[10]; const float* e1_b2 = (const float*)d_in[11];
  const float* g1_Wl = (const float*)d_in[12]; const float* g1_bl = (const float*)d_in[13];
  const float* g1_Wr = (const float*)d_in[14]; const float* g1_br = (const float*)d_in[15];
  const float* g1_We = (const float*)d_in[16]; const float* g1_att = (const float*)d_in[17];
  const float* g1_bias = (const float*)d_in[18];
  const float* bn_n_g = (const float*)d_in[19]; const float* bn_n_b = (const float*)d_in[20];
  const float* bn_e_g = (const float*)d_in[21]; const float* bn_e_b = (const float*)d_in[22];
  const float* e2_W1 = (const float*)d_in[23]; const float* e2_b1 = (const float*)d_in[24];
  const float* e2_W2 = (const float*)d_in[25]; const float* e2_b2 = (const float*)d_in[26];
  const float* g2_Wl = (const float*)d_in[27]; const float* g2_bl = (const float*)d_in[28];
  const float* g2_Wr = (const float*)d_in[29]; const float* g2_br = (const float*)d_in[30];
  const float* g2_We = (const float*)d_in[31]; const float* g2_att = (const float*)d_in[32];
  const float* g2_bias = (const float*)d_in[33];
  const float* fc1_W = (const float*)d_in[34]; const float* fc1_b = (const float*)d_in[35];
  const float* fc2_W = (const float*)d_in[36]; const float* fc2_b = (const float*)d_in[37];

  const int* src = eidx;
  const int* dst = eidx + EDGES;

  float* W = (float*)d_ws;
  float* h      = W;                         // N x 256
  float* h1     = W + 8388608;               // N x 256
  float* xl     = W + 16777216;              // N x 256
  float* xr     = W + 25165824;              // N x 256
  float* ehid   = W + 33554432;              // E x 128
  float* ea1    = W + 67108864;              // E x 128
  float* ea2    = W + 100663296;             // E x 128
  float* araw   = W + 134217728;             // (E+N) x 4
  float* aexp   = W + 135397376;             // (E+N) x 4
  float* amax   = W + 136577024;             // N x 4
  float* den    = W + 136708096;             // N x 4
  float* eamean = W + 136839168;             // 128
  float* mvn    = W + 136839296;             // 512
  float* mve    = W + 136839808;             // 256
  float* pooled = W + 136840064;             // G x 256
  float* fc1out = W + 136872832;             // G x 64
  bf16_t* PK    = (bf16_t*)(W + 136881024);  // packed bf16 weights (~1.1MB)
  bf16_t* p_ntW1 = PK;                       // 64x256    -> 16384
  bf16_t* p_ntW2 = PK + 16384;               // 256x256   -> 65536
  bf16_t* p_e1W1 = PK + 81920;               // 544x64    -> 34816
  bf16_t* p_e1W2 = PK + 116736;              // 64x128    -> 8192
  bf16_t* p_g1Wl = PK + 124928;              // 256x256
  bf16_t* p_g1Wr = PK + 190464;              // 256x256
  bf16_t* p_g1We = PK + 256000;              // 128x256   -> 32768
  bf16_t* p_e2W1 = PK + 288768;              // 640x128   -> 81920
  bf16_t* p_e2W2 = PK + 370688;              // 128x128   -> 16384
  bf16_t* p_g2Wl = PK + 387072;              // 256x256
  bf16_t* p_g2Wr = PK + 452608;              // 256x256
  bf16_t* p_g2We = PK + 518144;              // 128x256
  bf16_t* p_fc1  = PK + 550912;              // 256x64    -> 16384

  const int EN = EDGES + NODES;

  // ---- one-time weight packing (tiny, L2 resident) ----
  run_pack(nt_W1, 64, 256, p_ntW1, stream);
  run_pack(nt_W2, 256, 256, p_ntW2, stream);
  run_pack(e1_W1, 528, 64, p_e1W1, stream);
  run_pack(e1_W2, 64, 128, p_e1W2, stream);
  run_pack(g1_Wl, 256, 256, p_g1Wl, stream);
  run_pack(g1_Wr, 256, 256, p_g1Wr, stream);
  run_pack(g1_We, 128, 256, p_g1We, stream);
  run_pack(e2_W1, 640, 128, p_e2W1, stream);
  run_pack(e2_W2, 128, 128, p_e2W2, stream);
  run_pack(g2_Wl, 256, 256, p_g2Wl, stream);
  run_pack(g2_Wr, 256, 256, p_g2Wr, stream);
  run_pack(g2_We, 128, 256, p_g2We, stream);
  run_pack(fc1_W, 256, 64, p_fc1, stream);

  // ---- NodeTransform: h = relu(x@W1+b1)@W2+b2 (hidden staged in xl) ----
  run_gemm(x, 64, p_ntW1, nt_b1, xl, 256, NODES, 256, 64, 1, stream);
  run_gemm(xl, 256, p_ntW2, nt_b2, h, 256, NODES, 256, 256, 0, stream);

  // ---- layer1 edge MLP ----
  run_ecat(h, eattr0, 16, src, dst, p_e1W1, e1_b1, ehid, 64, EDGES, 64, 528, 1, stream);
  run_gemm(ehid, 64, p_e1W2, e1_b2, ea1, 128, EDGES, 128, 64, 0, stream);

  // ---- GAT layer 1 ----
  run_gemm(h, 256, p_g1Wl, g1_bl, xl, 256, NODES, 256, 256, 0, stream);
  run_gemm(h, 256, p_g1Wr, g1_br, xr, 256, NODES, 256, 256, 0, stream);
  fill(eamean, 0.f, 128, stream);
  col_mean<<<1024, 256, 0, stream>>>(ea1, EDGES, eamean);
  fill(amax, -1e30f, NODES * 4, stream);
  fill(den, 0.f, NODES * 4, stream);
  fill(h1, 0.f, NODES * 256, stream);
  gat_alpha_kernel<<<(EN / 16 + 7) / 8, 256, 0, stream>>>(
      ea1, eamean, p_g1We, xl, xr, src, dst, g1_att, araw, amax, EDGES, NODES);
  softmax_edge<<<(EN * 4 + 255) / 256, 256, 0, stream>>>(araw, amax, dst, aexp,
                                                         den, EDGES, NODES);
  scatter_msg<<<(EN * 256 + 255) / 256, 256, 0, stream>>>(aexp, den, xl, src,
                                                          dst, h1, EDGES, NODES);
  bias_relu<<<(NODES * 256 + 255) / 256, 256, 0, stream>>>(h1, g1_bias, NODES, 256);
  bn_stats<<<256, 256, 0, stream>>>(h1, NODES, 256, mvn);
  bn_apply<<<(NODES * 256 + 255) / 256, 256, 0, stream>>>(h1, NODES, 256, mvn,
                                                          bn_n_g, bn_n_b);
  bn_stats<<<128, 256, 0, stream>>>(ea1, EDGES, 128, mve);
  bn_apply<<<(EDGES * 128 + 255) / 256, 256, 0, stream>>>(ea1, EDGES, 128, mve,
                                                          bn_e_g, bn_e_b);

  // ---- layer2 edge MLP ----
  run_ecat(h1, ea1, 128, src, dst, p_e2W1, e2_b1, ehid, 128, EDGES, 128, 640, 1, stream);
  run_gemm(ehid, 128, p_e2W2, e2_b2, ea2, 128, EDGES, 128, 128, 0, stream);

  // ---- GAT layer 2 (accumulate into h, which is now free) ----
  run_gemm(h1, 256, p_g2Wl, g2_bl, xl, 256, NODES, 256, 256, 0, stream);
  run_gemm(h1, 256, p_g2Wr, g2_br, xr, 256, NODES, 256, 256, 0, stream);
  fill(eamean, 0.f, 128, stream);
  col_mean<<<1024, 256, 0, stream>>>(ea2, EDGES, eamean);
  fill(amax, -1e30f, NODES * 4, stream);
  fill(den, 0.f, NODES * 4, stream);
  fill(h, 0.f, NODES * 256, stream);
  gat_alpha_kernel<<<(EN / 16 + 7) / 8, 256, 0, stream>>>(
      ea2, eamean, p_g2We, xl, xr, src, dst, g2_att, araw, amax, EDGES, NODES);
  softmax_edge<<<(EN * 4 + 255) / 256, 256, 0, stream>>>(araw, amax, dst, aexp,
                                                         den, EDGES, NODES);
  scatter_msg<<<(EN * 256 + 255) / 256, 256, 0, stream>>>(aexp, den, xl, src,
                                                          dst, h, EDGES, NODES);
  bias_relu<<<(NODES * 256 + 255) / 256, 256, 0, stream>>>(h, g2_bias, NODES, 256);

  // ---- readout ----
  fill(pooled, 0.f, GR * 256, stream);
  pool_kernel<<<(NODES * 256 + 255) / 256, 256, 0, stream>>>(h, batch, pooled, NODES);
  run_gemm(pooled, 256, p_fc1, fc1_b, fc1out, 64, GR, 64, 256, 1, stream);
  fc2_kernel<<<1, GR, 0, stream>>>(fc1out, fc2_W, fc2_b, (float*)d_out, GR);
}
